// PointNetInteraction_88553635709092
// MI455X (gfx1250) — compile-verified, matrix operand order checked by reading, not run on hardware
//
#include <hip/hip_runtime.h>
#include <cstdint>
#include <cstddef>

// ---------------- types for WMMA (gfx1250, wave32) ----------------
typedef __attribute__((ext_vector_type(16))) __bf16          v16bf;
typedef __attribute__((ext_vector_type(8)))  float           v8f;
typedef __attribute__((ext_vector_type(8)))  unsigned short  us8;
typedef __attribute__((ext_vector_type(16))) unsigned short  us16;

__device__ __forceinline__ unsigned short f2bf(float f) {
  unsigned u = __float_as_uint(f);
  u += 0x7fffu + ((u >> 16) & 1u);          // round-to-nearest-even
  return (unsigned short)(u >> 16);
}
__device__ __forceinline__ float bf2f(unsigned short h) {
  return __uint_as_float(((unsigned)h) << 16);
}

// ---------------- gather coarse positions: pos_c = pos_f[idx] ----------------
__global__ void gather_pos(const float* __restrict__ src, const int* __restrict__ idx,
                           float* __restrict__ dst, int n) {
  int i = blockIdx.x * blockDim.x + threadIdx.x;
  if (i >= n * 3) return;
  int p = i / 3, c = i - p * 3;
  dst[i] = src[(size_t)idx[p] * 3 + c];
}

// ---------------- downsample edge-input builder ----------------
// Xin[e, :] = concat( x[src[e]] (D), pos_f[src[e]] - pos_c[dst[e]] (3), zeros ) in bf16
__global__ void build_down(const float* __restrict__ xsrc, int D,
                           const float* __restrict__ posf, const float* __restrict__ posc,
                           const int* __restrict__ esrc, const int* __restrict__ edst,
                           unsigned short* __restrict__ Xin, int Kpad) {
  int e = blockIdx.x;
  int s = esrc[e], d = edst[e];
  unsigned short* out = Xin + (size_t)e * Kpad;
  for (int j = threadIdx.x; j < Kpad; j += blockDim.x) {
    float v;
    if (j < D)          v = xsrc[(size_t)s * D + j];
    else if (j < D + 3) v = posf[(size_t)s * 3 + (j - D)] - posc[(size_t)d * 3 + (j - D)];
    else                v = 0.f;
    out[j] = f2bf(v);
  }
}

// ---------------- upsample (IDW interpolation) input builder ----------------
// Xin[f,:] = concat( sum_j w_j*xc[xidx_j] / sum_j w_j  (Dc), res1[f] (Dr1), res2[f] (Dr2), zeros )
__global__ void __launch_bounds__(128)
build_up(const float* __restrict__ xc, int Dc,
         const float* __restrict__ posc, const float* __restrict__ posf,
         const int* __restrict__ xidx, int k,
         const float* __restrict__ res1, int Dr1,
         const float* __restrict__ res2, int Dr2,
         unsigned short* __restrict__ Xin, int Kpad) {
  __shared__ int   sidx[32];
  __shared__ float sw[32];
  __shared__ float sden;
  int f = blockIdx.x, t = threadIdx.x;
  if (t < k) {
    int xi = xidx[(size_t)f * k + t];
    float dx = posc[xi * 3 + 0] - posf[f * 3 + 0];
    float dy = posc[xi * 3 + 1] - posf[f * 3 + 1];
    float dz = posc[xi * 3 + 2] - posf[f * 3 + 2];
    float d2 = dx * dx + dy * dy + dz * dz;
    if (d2 < 1e-16f) d2 = 1e-16f;
    sidx[t] = xi; sw[t] = 1.f / d2;
  }
  __syncthreads();
  if (t == 0) { float s = 0.f; for (int j = 0; j < k; ++j) s += sw[j]; sden = s; }
  __syncthreads();
  float inv_den = 1.f / sden;
  unsigned short* out = Xin + (size_t)f * Kpad;
  for (int j = t; j < Kpad; j += blockDim.x) {
    float v;
    if (j < Dc) {
      float acc = 0.f;
      for (int q = 0; q < k; ++q) acc += xc[(size_t)sidx[q] * Dc + j] * sw[q];
      v = acc * inv_den;
    } else if (j < Dc + Dr1)       v = res1[(size_t)f * Dr1 + (j - Dc)];
    else if (j < Dc + Dr1 + Dr2)   v = res2[(size_t)f * Dr2 + (j - Dc - Dr1)];
    else                           v = 0.f;
    out[j] = f2bf(v);
  }
}

// ---------------- pack fp32 W[K x N] -> bf16 WMMA B-fragment layout ----------------
// Wp[(kt*N + n)*32 + s]; s<16 -> lane-low half (K {0..7,16..23}); s>=16 -> lane-high (K {8..15,24..31})
__global__ void pack_w(const float* __restrict__ W, int K, int N, int Kpad,
                       unsigned short* __restrict__ Wp) {
  int i = blockIdx.x * blockDim.x + threadIdx.x;
  int total = (Kpad >> 5) * N * 32;
  if (i >= total) return;
  int s   = i & 31;
  int rem = i >> 5;
  int n   = rem % N;
  int kt  = rem / N;
  int e   = (s < 16) ? s : s - 16;
  int kin = (e < 8 ? e : e + 8) + (s < 16 ? 0 : 8);
  int kk  = kt * 32 + kin;
  float v = (kk < K) ? W[(size_t)kk * N + n] : 0.f;
  Wp[i] = f2bf(v);
}

// ---------------- GEMM: out = silu(A @ W + b), bf16 in, fp32 accum ----------------
// Register-blocked: each wave owns one 16-row M tile and FOUR consecutive 16-col N tiles,
// so the A fragment is loaded once per K step and reused 4x (A traffic /4).
__global__ void __launch_bounds__(256)
gemm_silu_wmma(const unsigned short* __restrict__ A,   // [M x Kpad] bf16 row-major
               const unsigned short* __restrict__ Wp,  // packed fragments
               const float* __restrict__ bias,
               unsigned short* __restrict__ Obf, float* __restrict__ Of,
               int M, int Kpad, int N, int out_f32) {
  const int lane = threadIdx.x & 31;
  const int wv   = threadIdx.x >> 5;
  const int mt   = blockIdx.x * 8 + wv;
  const int nt0  = blockIdx.y * 4;          // 4 consecutive n-tiles per wave
  if (mt * 16 >= M) return;                 // wave-uniform: EXEC stays all-ones
  const int ll = lane & 15;
  const int hi = lane >> 4;
  const unsigned short* Arow = A + (size_t)(mt * 16 + ll) * Kpad + hi * 8;
  v8f acc0 = {}, acc1 = {}, acc2 = {}, acc3 = {};
  const int ktiles = Kpad >> 5;
  const unsigned short* Bcol = Wp + (size_t)(nt0 * 16 + ll) * 32 + hi * 16;
  for (int kt = 0; kt < ktiles; ++kt) {
    union { us16 u; v16bf b; us8 h[2]; } fa, fb0, fb1, fb2, fb3;
    fa.h[0] = *(const us8*)(Arow + kt * 32);
    fa.h[1] = *(const us8*)(Arow + kt * 32 + 16);
    const unsigned short* Bk = Bcol + (size_t)kt * N * 32;
    fb0.u = *(const us16*)(Bk);
    fb1.u = *(const us16*)(Bk + 16 * 32);
    fb2.u = *(const us16*)(Bk + 32 * 32);
    fb3.u = *(const us16*)(Bk + 48 * 32);
    acc0 = __builtin_amdgcn_wmma_f32_16x16x32_bf16(false, fa.b, false, fb0.b,
                                                   (short)0, acc0, false, false);
    acc1 = __builtin_amdgcn_wmma_f32_16x16x32_bf16(false, fa.b, false, fb1.b,
                                                   (short)0, acc1, false, false);
    acc2 = __builtin_amdgcn_wmma_f32_16x16x32_bf16(false, fa.b, false, fb2.b,
                                                   (short)0, acc2, false, false);
    acc3 = __builtin_amdgcn_wmma_f32_16x16x32_bf16(false, fa.b, false, fb3.b,
                                                   (short)0, acc3, false, false);
  }
  const int rbase = mt * 16 + hi * 8;
  v8f accs[4] = {acc0, acc1, acc2, acc3};
#pragma unroll
  for (int j = 0; j < 4; ++j) {
    const int n = (nt0 + j) * 16 + ll;
    const float bsc = bias[n];
#pragma unroll
    for (int i = 0; i < 8; ++i) {
      float x = accs[j][i] + bsc;
      float s = x / (1.f + __expf(-x));     // SiLU
      size_t o = (size_t)(rbase + i) * N + n;
      if (out_f32) Of[o] = s; else Obf[o] = f2bf(s);
    }
  }
}

// ---------------- block reduction helper (blockDim == 128) ----------------
__device__ __forceinline__ float block_sum(float v, float* red) {
  int t = threadIdx.x;
  red[t] = v; __syncthreads();
  for (int s = 64; s > 0; s >>= 1) { if (t < s) red[t] += red[t + s]; __syncthreads(); }
  float r = red[0]; __syncthreads();
  return r;
}

// ---------------- LayerNorm + mean over k consecutive rows (downsample) ----------------
__global__ void __launch_bounds__(128)
ln_mean(const unsigned short* __restrict__ F, int D, int k,
        const float* __restrict__ g, const float* __restrict__ beta,
        float* __restrict__ out) {
  __shared__ float red[128];
  const int c = blockIdx.x, t = threadIdx.x;
  const int per = (D + 127) >> 7;
  float acc[4] = {0.f, 0.f, 0.f, 0.f};
  float v[4];
  for (int r = 0; r < k; ++r) {
    const unsigned short* row = F + (size_t)(c * k + r) * D;
    float s = 0.f;
    for (int p = 0; p < per; ++p) {
      int col = t + (p << 7);
      float x = (col < D) ? bf2f(row[col]) : 0.f;
      v[p] = x; s += x;
    }
    float mu = block_sum(s, red) / (float)D;
    float q = 0.f;
    for (int p = 0; p < per; ++p) {
      int col = t + (p << 7);
      if (col < D) { float dd = v[p] - mu; q += dd * dd; }
    }
    float var  = block_sum(q, red) / (float)D;
    float rstd = rsqrtf(var + 1e-5f);
    for (int p = 0; p < per; ++p) {
      int col = t + (p << 7);
      if (col < D) acc[p] += (v[p] - mu) * rstd * g[col] + beta[col];
    }
  }
  float inv = 1.f / (float)k;
  for (int p = 0; p < per; ++p) {
    int col = t + (p << 7);
    if (col < D) out[(size_t)c * D + col] = acc[p] * inv;
  }
}

// ---------------- LayerNorm only (upsample / final output) ----------------
__global__ void __launch_bounds__(128)
ln_rows(const unsigned short* __restrict__ F, int D,
        const float* __restrict__ g, const float* __restrict__ beta,
        float* __restrict__ out) {
  __shared__ float red[128];
  const int r = blockIdx.x, t = threadIdx.x;
  const int per = (D + 127) >> 7;
  const unsigned short* row = F + (size_t)r * D;
  float v[4]; float s = 0.f;
  for (int p = 0; p < per; ++p) {
    int col = t + (p << 7);
    float x = (col < D) ? bf2f(row[col]) : 0.f;
    v[p] = x; s += x;
  }
  float mu = block_sum(s, red) / (float)D;
  float q = 0.f;
  for (int p = 0; p < per; ++p) {
    int col = t + (p << 7);
    if (col < D) { float dd = v[p] - mu; q += dd * dd; }
  }
  float var  = block_sum(q, red) / (float)D;
  float rstd = rsqrtf(var + 1e-5f);
  for (int p = 0; p < per; ++p) {
    int col = t + (p << 7);
    if (col < D) out[(size_t)r * D + col] = (v[p] - mu) * rstd * g[col] + beta[col];
  }
}

// ---------------- host orchestration ----------------
extern "C" void kernel_launch(void* const* d_in, const int* in_sizes, int n_in,
                              void* d_out, int out_size, void* d_ws, size_t ws_size,
                              hipStream_t stream) {
  (void)in_sizes; (void)n_in; (void)out_size; (void)ws_size;
  const int N0 = 32768, N1 = 16384, N2 = 4096, N3 = 512;
  const int E1 = N1 * 32, E2 = N2 * 32, E3 = N3 * 32;

  const float* nodes  = (const float*)d_in[0];
  const float* coords = (const float*)d_in[1];
  // params flattened (sorted dict): down1,down2,down3,up1,up2,up3; each w0,b0,w1,b1,w2,b2,g,beta
  const float* P[48];
  for (int i = 0; i < 48; ++i) P[i] = (const float*)d_in[2 + i];
  const int* idx1 = (const int*)d_in[50];
  const int* idx2 = (const int*)d_in[51];
  const int* idx3 = (const int*)d_in[52];
  const int* e1   = (const int*)d_in[53];
  const int* e2   = (const int*)d_in[54];
  const int* e3   = (const int*)d_in[55];
  const int* i3   = (const int*)d_in[56];
  const int* i2   = (const int*)d_in[57];
  const int* i1   = (const int*)d_in[58];

  // ---- workspace carve-out ----
  char* wsp = (char*)d_ws;
  auto alloc = [&](size_t b) -> void* {
    void* p = (void*)wsp; wsp += (b + 255) & ~(size_t)255; return p;
  };
  float* pos1 = (float*)alloc((size_t)N1 * 3 * 4);
  float* pos2 = (float*)alloc((size_t)N2 * 3 * 4);
  float* pos3 = (float*)alloc((size_t)N3 * 3 * 4);
  float* x1   = (float*)alloc((size_t)N1 * 128 * 4);
  float* x2   = (float*)alloc((size_t)N2 * 256 * 4);
  float* x3   = (float*)alloc((size_t)N3 * 512 * 4);
  float* d3   = (float*)alloc((size_t)N2 * 256 * 4);
  float* d2   = (float*)alloc((size_t)N1 * 128 * 4);
  unsigned short* Wp   = (unsigned short*)alloc((size_t)768 * 512 * 2);
  unsigned short* regA = (unsigned short*)alloc((size_t)E1 * 128 * 2); // Xin and pre-LN F (aliased)
  unsigned short* regB = (unsigned short*)alloc((size_t)E1 * 64 * 2);  // hidden 1
  unsigned short* regC = (unsigned short*)alloc((size_t)E1 * 64 * 2);  // hidden 2

  auto gemm = [&](const unsigned short* A, const float* W, const float* b,
                  unsigned short* Obf, float* Of, int M, int K, int Kp, int N, int outf) {
    int tot = (Kp >> 5) * N * 32;
    pack_w<<<dim3((tot + 255) / 256), dim3(256), 0, stream>>>(W, K, N, Kp, Wp);
    gemm_silu_wmma<<<dim3((M / 16 + 7) / 8, N / 64), dim3(256), 0, stream>>>(
        A, Wp, b, Obf, Of, M, Kp, N, outf);
  };
  // three-layer silu MLP (all layers end in silu; LN handled by caller)
  auto run_mlp = [&](int M, int K0, int K0p, int H, int Nout, const float* const* pp) {
    gemm(regA, pp[0], pp[1], regB, nullptr, M, K0, K0p, H, 0);
    gemm(regB, pp[2], pp[3], regC, nullptr, M, H, H, H, 0);
    gemm(regC, pp[4], pp[5], regA, nullptr, M, H, H, Nout, 0); // regA's Xin is dead by now
  };

  // ---- coarse positions ----
  gather_pos<<<dim3((N1 * 3 + 255) / 256), dim3(256), 0, stream>>>(coords, idx1, pos1, N1);
  gather_pos<<<dim3((N2 * 3 + 255) / 256), dim3(256), 0, stream>>>(pos1,   idx2, pos2, N2);
  gather_pos<<<dim3((N3 * 3 + 255) / 256), dim3(256), 0, stream>>>(pos2,   idx3, pos3, N3);

  // ---- stage A: up1 (P[24..31])  35 -> 64 -> 64 -> 128, mean over 32 ----
  build_down<<<dim3(E1), dim3(64), 0, stream>>>(nodes, 32, coords, pos1, e1, e1 + E1, regA, 64);
  run_mlp(E1, 35, 64, 64, 128, &P[24]);
  ln_mean<<<dim3(N1), dim3(128), 0, stream>>>(regA, 128, 32, P[30], P[31], x1);

  // ---- stage B: up2 (P[32..39])  131 -> 128 -> 128 -> 256 ----
  build_down<<<dim3(E2), dim3(64), 0, stream>>>(x1, 128, pos1, pos2, e2, e2 + E2, regA, 160);
  run_mlp(E2, 131, 160, 128, 256, &P[32]);
  ln_mean<<<dim3(N2), dim3(128), 0, stream>>>(regA, 256, 32, P[38], P[39], x2);

  // ---- stage C: up3 (P[40..47])  259 -> 256 -> 256 -> 512 ----
  build_down<<<dim3(E3), dim3(64), 0, stream>>>(x2, 256, pos2, pos3, e3, e3 + E3, regA, 288);
  run_mlp(E3, 259, 288, 256, 512, &P[40]);
  ln_mean<<<dim3(N3), dim3(128), 0, stream>>>(regA, 512, 32, P[46], P[47], x3);

  // ---- stage D: down3 (P[16..23])  interp(x3,k=32)||x2 : 768 -> 256 -> 256 -> 256 ----
  build_up<<<dim3(N2), dim3(128), 0, stream>>>(x3, 512, pos3, pos2, i3, 32,
                                               x2, 256, nullptr, 0, regA, 768);
  run_mlp(N2, 768, 768, 256, 256, &P[16]);
  ln_rows<<<dim3(N2), dim3(128), 0, stream>>>(regA, 256, P[22], P[23], d3);

  // ---- stage E: down2 (P[8..15])  interp(d3,k=16)||x1 : 384 -> 128 -> 128 -> 128 ----
  build_up<<<dim3(N1), dim3(128), 0, stream>>>(d3, 256, pos2, pos1, i2, 16,
                                               x1, 128, nullptr, 0, regA, 384);
  run_mlp(N1, 384, 384, 128, 128, &P[8]);
  ln_rows<<<dim3(N1), dim3(128), 0, stream>>>(regA, 128, P[14], P[15], d2);

  // ---- stage F: down1 (P[0..7])  interp(d2,k=8)||nodes||coords : 163 -> 64 -> 64 -> 64 ----
  build_up<<<dim3(N0), dim3(128), 0, stream>>>(d2, 128, pos1, coords, i1, 8,
                                               nodes, 32, coords, 3, regA, 192);
  run_mlp(N0, 163, 192, 64, 64, &P[0]);
  ln_rows<<<dim3(N0), dim3(128), 0, stream>>>(regA, 64, P[6], P[7], (float*)d_out);
}
